// LSS_72782515798143
// MI455X (gfx1250) — compile-verified, matrix operand order checked by reading, not run on hardware
//
#include <hip/hip_runtime.h>
#include <hip/hip_bf16.h>

typedef __attribute__((ext_vector_type(16))) _Float16 v16h;
typedef __attribute__((ext_vector_type(8)))  float    v8f;

namespace {
constexpr int kB  = 8,  kN  = 6,  kD = 45, kC = 64, kFH = 8, kFW = 22;
constexpr int kGX = 64, kGY = 64, kGZ = 32;
constexpr int kSpat = kGZ * kGX * kGY;     // 131072 floats per (b,c) slab
constexpr int kHW   = kFH * kFW;           // 176
constexpr int kPix  = kB * kN * kFH * kFW; // 8448 rays -> one wave32 each
}

// ---------------------------------------------------------------- zero init
__global__ void lss_zero_kernel(float4* __restrict__ out, int n4) {
  int i = blockIdx.x * blockDim.x + threadIdx.x;
  int stride = gridDim.x * blockDim.x;
  float4 z; z.x = z.y = z.z = z.w = 0.0f;
  for (; i < n4; i += stride) out[i] = z;
}

// ---------------------------------------------- per-camera M = R*inv(K), t
__global__ void lss_geom_kernel(const float* __restrict__ rots,
                                const float* __restrict__ trans,
                                const float* __restrict__ intr,
                                float* __restrict__ Mt) {
  int i = blockIdx.x * blockDim.x + threadIdx.x;   // i = b*N + n
  if (i >= kB * kN) return;
  const float* K = intr + i * 9;
  const float* R = rots + i * 9;
  float a = K[0], b = K[1], c = K[2];
  float d = K[3], e = K[4], f = K[5];
  float g = K[6], h = K[7], ii = K[8];
  float c00 =  (e * ii - f * h);
  float c10 = -(d * ii - f * g);
  float c20 =  (d * h  - e * g);
  float inv = 1.0f / (a * c00 + b * c10 + c * c20);
  float iK[9];
  iK[0] = c00 * inv;            iK[1] = (c * h - b * ii) * inv;  iK[2] = (b * f - c * e) * inv;
  iK[3] = c10 * inv;            iK[4] = (a * ii - c * g) * inv;  iK[5] = (c * d - a * f) * inv;
  iK[6] = c20 * inv;            iK[7] = (b * g - a * h) * inv;   iK[8] = (a * e - b * d) * inv;
  float* M = Mt + i * 12;
#pragma unroll
  for (int r = 0; r < 3; ++r)
#pragma unroll
    for (int cc = 0; cc < 3; ++cc)
      M[r * 3 + cc] = R[r * 3 + 0] * iK[0 * 3 + cc] +
                      R[r * 3 + 1] * iK[1 * 3 + cc] +
                      R[r * 3 + 2] * iK[2 * 3 + cc];
  M[9]  = trans[i * 3 + 0];
  M[10] = trans[i * 3 + 1];
  M[11] = trans[i * 3 + 2];
}

// voxel index (or -1) for depth bin dd on ray (v,t); matches reference:
// nx from g.x, ny from g.z, nz from g.y; truncation toward zero.
__device__ __forceinline__ int voxel_index(int dd, float vx, float vy, float vz,
                                           float tx, float ty, float tz, int b) {
  if (dd >= kD) return -1;
  float dv = (float)(dd + 1);
  float gx = dv * vx + tx;
  float gy = dv * vy + ty;
  float gz = dv * vz + tz;
  int nx = (int)((gx + 16.0f) * 2.0f);
  int ny = (int)((gz + 16.0f) * 2.0f);
  int nz = (int)((gy + 8.0f)  * 2.0f);
  if (nx < 0 || nx >= kGX || ny < 0 || ny >= kGY || nz < 0 || nz >= kGZ) return -1;
  return b * (kC * kSpat) + nz * (kGX * kGY) + nx * kGY + ny;
}

// ------------------------------------------------------------- main splat
// One wave32 per (b,n,h,w) ray. ctx tile (64x48) = feature (x) depth via 12
// v_wmma_f32_16x16x32_f16 ops (K=0 slice carries the vectors). A rows = 64
// channels, B cols = 48 depth bins, so the accumulator's N dimension is the
// depth bin: each lane's 8 acc elements share ONE voxel -> one predicate and
// one address register pair per 8 global_atomic_add_f32 (channel offsets
// v*kSpat*4 <= 3.67 MB fold into the 24-bit instruction offset).
__global__ void __launch_bounds__(256) lss_splat_kernel(
    const float* __restrict__ depth,
    const float* __restrict__ feature,
    const float* __restrict__ Mt,
    float* __restrict__ out) {
  const int lane = threadIdx.x & 31;
  const int wave = blockIdx.x * (blockDim.x >> 5) + (threadIdx.x >> 5);
  if (wave >= kPix) return;                        // uniform per wave

  int w  = wave % kFW;
  int t1 = wave / kFW;
  int h  = t1 % kFH;
  int bn = t1 / kFH;                               // b*N + n
  int b  = bn / kN;

  const float* M = Mt + bn * 12;
  float px = (float)w * (351.0f / 21.0f);
  float py = (float)h * (127.0f / 7.0f);
  float vx = M[0] * px + M[1] * py + M[2];
  float vy = M[3] * px + M[4] * py + M[5];
  float vz = M[6] * px + M[7] * py + M[8];
  float tx = M[9], ty = M[10], tz = M[11];

  // lane-parallel voxel precompute: lane owns bins lane and lane+32
  int vox0 = voxel_index(lane,      vx, vy, vz, tx, ty, tz, b);
  int vox1 = voxel_index(lane + 32, vx, vy, vz, tx, ty, tz, b);

  // lane-parallel strided loads: depth[b,n,d,h,w], feature[b,n,c,h,w]
  const float* dptr = depth   + (bn * kD) * kHW + h * kFW + w;
  const float* fptr = feature + (bn * kC) * kHW + h * kFW + w;
  float dep0 = dptr[lane * kHW];                                   // d = lane
  float dep1 = (lane + 32 < kD) ? dptr[(lane + 32) * kHW] : 0.0f;  // d = lane+32
  float f0   = fptr[lane * kHW];                                   // c = lane
  float f1   = fptr[(lane + 32) * kHW];                            // c = lane+32

  const int  half = lane & 15;
  const bool lo   = lane < 16;
  const int  doff = lo ? 0 : 8;                    // hi half handles M = v+8

  // A = feature (M = channel): 4 tiles of 16 channels, prebuilt once.
  // Only K=0 is nonzero -> VGPR0 low half of lanes 0..15.
  _Float16 acol[4];
  acol[0] = (_Float16)__shfl(f0, half,      32);   // c = 0..15
  acol[1] = (_Float16)__shfl(f0, half + 16, 32);   // c = 16..31
  acol[2] = (_Float16)__shfl(f1, half,      32);   // c = 32..47
  acol[3] = (_Float16)__shfl(f1, half + 16, 32);   // c = 48..63
  v16h Afrag[4];
#pragma unroll
  for (int ct = 0; ct < 4; ++ct) {
    v16h t = {};
    t[0] = lo ? acol[ct] : (_Float16)0.0f;
    Afrag[ct] = t;
  }

  // B = depth (N = depth bin): 3 tiles of 16 bins.
  _Float16 drow[3];
  drow[0] = (_Float16)__shfl(dep0, half,      32); // d = 0..15
  drow[1] = (_Float16)__shfl(dep0, half + 16, 32); // d = 16..31
  drow[2] = (_Float16)__shfl(dep1, half,      32); // d = 32..47 (>=45 are 0)

#pragma unroll
  for (int dt = 0; dt < 3; ++dt) {
    v16h Bm = {};
    Bm[0] = lo ? drow[dt] : (_Float16)0.0f;

    // this lane's depth bin for this tile: d = dt*16 + half -> one voxel
    int vox = (dt == 0) ? __shfl(vox0, half,      32)
            : (dt == 1) ? __shfl(vox0, half + 16, 32)
                        : __shfl(vox1, half,      32);

#pragma unroll
    for (int ct = 0; ct < 4; ++ct) {
      v8f acc = {};
      acc = __builtin_amdgcn_wmma_f32_16x16x32_f16(
          /*neg_a=*/false, Afrag[ct], /*neg_b=*/false, Bm,
          /*c_mod=*/(short)0, acc, /*reuse_a=*/false, /*reuse_b=*/false);

      // acc[v] belongs to channel c = ct*16 + doff + v, depth bin d (fixed).
      if (vox >= 0) {
        float* p = out + vox + (ct * 16 + doff) * kSpat;
#pragma unroll
        for (int v = 0; v < 8; ++v)
          atomicAdd(p + v * kSpat, acc[v]);   // v*kSpat*4 -> inst offset
      }
    }
  }
}

// ------------------------------------------------------------------ launch
extern "C" void kernel_launch(void* const* d_in, const int* in_sizes, int n_in,
                              void* d_out, int out_size, void* d_ws, size_t ws_size,
                              hipStream_t stream) {
  const float* depth   = (const float*)d_in[0];
  const float* feature = (const float*)d_in[1];
  const float* rots    = (const float*)d_in[2];
  const float* trans   = (const float*)d_in[3];
  const float* intr    = (const float*)d_in[4];
  float* out = (float*)d_out;
  float* Mt  = (float*)d_ws;          // 48 * 12 floats

  // 1) zero 256 MB output with float4 stores (~11 us at HBM peak)
  int n4 = out_size / 4;
  lss_zero_kernel<<<2048, 256, 0, stream>>>((float4*)out, n4);

  // 2) per-camera M = R * inv(K) and translation
  lss_geom_kernel<<<1, 64, 0, stream>>>(rots, trans, intr, Mt);

  // 3) one wave per ray: 8448 waves = 1056 blocks x 8 waves
  lss_splat_kernel<<<kPix / 8, 256, 0, stream>>>(depth, feature, Mt, out);
}
